// GraphMemoryVQ_1563368096467
// MI455X (gfx1250) — matrix operand order repaired and checked by reading, not compiled, not run on hardware
//
#include <hip/hip_runtime.h>
#include <stdint.h>

#define N_TOKENS   65536
#define LATENT_DIM 256
#define D2         512            // 2*LATENT_DIM
#define N_SYMBOLS  4096

typedef __attribute__((ext_vector_type(16))) __bf16 bf16x16;
typedef __attribute__((ext_vector_type(8)))  float  f32x8;

union BFrag { bf16x16 v; uint4 q[2]; };

// ---- output layout (floats): [z_complex interleaved | loss | min_idx | perplexity]
#define Z_OUT_SZ   ((size_t)N_TOKENS * D2)       // 33554432
#define LOSS_OFF   (Z_OUT_SZ)
#define IDX_OFF    (Z_OUT_SZ + 1)
#define PERP_OFF   (Z_OUT_SZ + 1 + N_TOKENS)

// ---- workspace layout (bytes)
#define WS_MINIDX  0x000000     // int[65536]
#define WS_COUNTS  0x040000     // uint[4096]
#define WS_LOSS    0x044000     // double
#define WS_C2      0x045000     // float[4096]
#define WS_CBH     0x049000     // bf16[4096*512]  (4 MB)
#define WS_CBL     0x449000     // bf16[4096*512]  (4 MB)

// ---------------------------------------------------------------------------
// Kernel 0: codebook -> bf16 hi/lo split + ||c||^2
// ---------------------------------------------------------------------------
__global__ __launch_bounds__(128)
void cb_prep(const float* __restrict__ cb, uint16_t* __restrict__ cbh_u,
             uint16_t* __restrict__ cbl_u, float* __restrict__ c2) {
    __bf16* cbh = (__bf16*)cbh_u;
    __bf16* cbl = (__bf16*)cbl_u;
    const int row = blockIdx.x;
    const int tid = threadIdx.x;
    float s = 0.f;
    for (int k = tid; k < D2; k += 128) {
        float v = cb[(size_t)row * D2 + k];
        __bf16 h = (__bf16)v;
        cbh[(size_t)row * D2 + k] = h;
        cbl[(size_t)row * D2 + k] = (__bf16)(v - (float)h);
        s += v * v;
    }
    #pragma unroll
    for (int m = 16; m >= 1; m >>= 1) s += __shfl_xor(s, m, 32);
    __shared__ float red[4];
    if ((tid & 31) == 0) red[tid >> 5] = s;
    __syncthreads();
    if (tid == 0) c2[row] = red[0] + red[1] + red[2] + red[3];
}

// ---------------------------------------------------------------------------
// Kernel 1: WMMA distance + fused bias + running argmin
//   block = 128 threads (4 waves), 64 tokens/block, wave w -> tokens w*16..+15
// ---------------------------------------------------------------------------
__global__ __launch_bounds__(128)
void vq_argmin(const float* __restrict__ zr, const float* __restrict__ zi,
               const int* __restrict__ prev,
               const uint16_t* __restrict__ cbh_u, const uint16_t* __restrict__ cbl_u,
               const float* __restrict__ c2, const float* __restrict__ adj,
               int* __restrict__ min_idx) {
    const __bf16* cbh = (const __bf16*)cbh_u;
    const __bf16* cbl = (const __bf16*)cbl_u;

    // padded row stride 520 (= 1040B, multiple of 16B; 4-bank rotation per row)
    __shared__ __bf16 zh[64][520];
    __shared__ __bf16 zl[64][520];
    __shared__ int    prevRow[64];

    const int tid  = threadIdx.x;
    const int tok0 = blockIdx.x * 64;

    // Phase 0: convert this block's 64 z rows to bf16 hi/lo in LDS
    for (int i = tid; i < 64 * D2; i += 128) {
        int t = i >> 9, k = i & (D2 - 1);
        float v = (k < LATENT_DIM) ? zr[(size_t)(tok0 + t) * LATENT_DIM + k]
                                   : zi[(size_t)(tok0 + t) * LATENT_DIM + (k - LATENT_DIM)];
        __bf16 h = (__bf16)v;
        zh[t][k] = h;
        zl[t][k] = (__bf16)(v - (float)h);
    }
    if (tid < 64) prevRow[tid] = prev[tok0 + tid];
    __syncthreads();

    const int wave  = tid >> 5;
    const int lane  = tid & 31;
    const int half  = lane >> 4;     // 0: lanes 0-15, 1: lanes 16-31
    const int mrow  = lane & 15;     // A row / B col within tile
    const int mbase = wave * 16;
    const int tbase = mbase + half * 8;  // C-fragment: token = tbase + v

    // cache adjacency row bases for the 8 tokens this lane's C covers
    const float* arow[8];
    #pragma unroll
    for (int v = 0; v < 8; v++)
        arow[v] = adj + (size_t)prevRow[tbase + v] * N_SYMBOLS;

    const __bf16* zrow_h = &zh[mbase + mrow][0];
    const __bf16* zrow_l = &zl[mbase + mrow][0];

    float bestd[8];
    int   besti[8];
    #pragma unroll
    for (int v = 0; v < 8; v++) { bestd[v] = 3.402823466e38f; besti[v] = 0; }

    for (int ntile = 0; ntile < N_SYMBOLS / 16; ++ntile) {
        const int nGlob = ntile * 16 + mrow;
        const __bf16* bhrow = cbh + (size_t)nGlob * D2;
        const __bf16* blrow = cbl + (size_t)nGlob * D2;
        if (ntile + 1 < N_SYMBOLS / 16) {          // prefetch next tile's rows
            __builtin_prefetch(bhrow + 16 * D2, 0, 0);
            __builtin_prefetch(blrow + 16 * D2, 0, 0);
        }

        f32x8 acc = {};
        #pragma unroll
        for (int kb = 0; kb < D2; kb += 32) {
            // A (16x32 bf16): lane holds K = kb + half*8 + {0..7} and +16
            const int kkA = kb + half * 8;
            BFrag ah, al, bh, bl;
            ah.q[0] = *(const uint4*)(zrow_h + kkA);
            ah.q[1] = *(const uint4*)(zrow_h + kkA + 16);
            al.q[0] = *(const uint4*)(zrow_l + kkA);
            al.q[1] = *(const uint4*)(zrow_l + kkA + 16);
            // B (32x16 bf16): lane holds K = kb + half*8 + {0..7} and +16
            const int kkB = kb + half * 8;
            bh.q[0] = *(const uint4*)(bhrow + kkB);
            bh.q[1] = *(const uint4*)(bhrow + kkB + 16);
            bl.q[0] = *(const uint4*)(blrow + kkB);
            bl.q[1] = *(const uint4*)(blrow + kkB + 16);

            // dot = hi*hi + hi*lo + lo*hi  (fp32 accumulate; ~2^-18 rel err)
            acc = __builtin_amdgcn_wmma_f32_16x16x32_bf16(false, ah.v, false, bh.v,
                                                          (short)0, acc, false, false);
            acc = __builtin_amdgcn_wmma_f32_16x16x32_bf16(false, ah.v, false, bl.v,
                                                          (short)0, acc, false, false);
            acc = __builtin_amdgcn_wmma_f32_16x16x32_bf16(false, al.v, false, bh.v,
                                                          (short)0, acc, false, false);
        }

        const float c2v = c2[nGlob];
        #pragma unroll
        for (int v = 0; v < 8; v++) {
            float a  = arow[v][nGlob];
            float sg = 0.5f / (1.0f + __expf(-a));       // 0.5 * sigmoid
            float d  = c2v - 2.0f * acc[v] - sg;          // ||z||^2 dropped (const per token)
            if (d < bestd[v]) { bestd[v] = d; besti[v] = nGlob; }
        }
    }

    // reduce over the 16 lanes of this half (each lane saw n % 16 == mrow)
    #pragma unroll
    for (int v = 0; v < 8; v++) {
        float d = bestd[v];
        int   i = besti[v];
        #pragma unroll
        for (int m = 1; m < 16; m <<= 1) {
            float od = __shfl_xor(d, m, 32);
            int   oi = __shfl_xor(i, m, 32);
            if (od < d || (od == d && oi < i)) { d = od; i = oi; }
        }
        if (mrow == 0) min_idx[tok0 + tbase + v] = i;
    }
}

// ---------------------------------------------------------------------------
// Kernel 2: gather z_q, write interleaved complex out, loss partials, counts
//   one wave per token, 8 tokens per 256-thread block
// ---------------------------------------------------------------------------
__device__ __forceinline__ float sq4(float4 a, float4 b) {
    float dx = a.x - b.x, dy = a.y - b.y, dz = a.z - b.z, dw = a.w - b.w;
    return dx * dx + dy * dy + dz * dz + dw * dw;
}

__global__ __launch_bounds__(256)
void vq_output(const float* __restrict__ zr, const float* __restrict__ zi,
               const float* __restrict__ cb, const int* __restrict__ min_idx,
               float* __restrict__ out, double* __restrict__ loss_accum,
               unsigned* __restrict__ counts) {
    const int lane = threadIdx.x & 31;
    const int w    = threadIdx.x >> 5;
    const int tok  = blockIdx.x * 8 + w;
    const int idx  = min_idx[tok];
    const float* crow = cb + (size_t)idx * D2;
    const float* zrp  = zr + (size_t)tok * LATENT_DIM;
    const float* zip  = zi + (size_t)tok * LATENT_DIM;

    const int k0 = lane * 8;   // 8 real dims + 8 imag dims per lane
    float4 cr0 = *(const float4*)(crow + k0);
    float4 cr1 = *(const float4*)(crow + k0 + 4);
    float4 ci0 = *(const float4*)(crow + LATENT_DIM + k0);
    float4 ci1 = *(const float4*)(crow + LATENT_DIM + k0 + 4);
    float4 a0  = *(const float4*)(zrp + k0);
    float4 a1  = *(const float4*)(zrp + k0 + 4);
    float4 b0  = *(const float4*)(zip + k0);
    float4 b1  = *(const float4*)(zip + k0 + 4);

    float s = sq4(cr0, a0) + sq4(cr1, a1) + sq4(ci0, b0) + sq4(ci1, b1);

    // interleaved complex store: out[(tok*256 + k)*2 + {0,1}] = {re, im}
    float4* ob = (float4*)(out + ((size_t)tok * LATENT_DIM + k0) * 2);
    ob[0] = make_float4(cr0.x, ci0.x, cr0.y, ci0.y);
    ob[1] = make_float4(cr0.z, ci0.z, cr0.w, ci0.w);
    ob[2] = make_float4(cr1.x, ci1.x, cr1.y, ci1.y);
    ob[3] = make_float4(cr1.z, ci1.z, cr1.w, ci1.w);

    #pragma unroll
    for (int m = 16; m >= 1; m >>= 1) s += __shfl_xor(s, m, 32);

    __shared__ float wsum[8];
    if (lane == 0) {
        wsum[w] = s;
        atomicAdd(&counts[idx], 1u);
        out[IDX_OFF + tok] = (float)idx;
    }
    __syncthreads();
    if (threadIdx.x == 0) {
        float t = 0.f;
        #pragma unroll
        for (int j = 0; j < 8; j++) t += wsum[j];
        atomicAdd(loss_accum, (double)t);    // global_atomic_add_f64
    }
}

// ---------------------------------------------------------------------------
// Kernel 3: finalize loss + perplexity
// ---------------------------------------------------------------------------
__global__ __launch_bounds__(256)
void vq_finalize(const double* __restrict__ loss_accum,
                 const unsigned* __restrict__ counts, float* __restrict__ out) {
    const int tid = threadIdx.x;
    double s = 0.0;
    for (int n = tid; n < N_SYMBOLS; n += 256) {
        double p = (double)counts[n] * (1.0 / (double)N_TOKENS);
        s += p * log(p + 1e-10);
    }
    #pragma unroll
    for (int m = 16; m >= 1; m >>= 1) s += __shfl_xor(s, m, 32);
    __shared__ double red[8];
    if ((tid & 31) == 0) red[tid >> 5] = s;
    __syncthreads();
    if (tid == 0) {
        double tot = 0.0;
        #pragma unroll
        for (int j = 0; j < 8; j++) tot += red[j];
        out[PERP_OFF] = (float)exp(-tot);
        // loss_vq == loss_commit numerically -> loss = 1.25 * mean
        out[LOSS_OFF] = (float)(1.25 * (*loss_accum) / ((double)N_TOKENS * (double)D2));
    }
}

// ---------------------------------------------------------------------------
extern "C" void kernel_launch(void* const* d_in, const int* in_sizes, int n_in,
                              void* d_out, int out_size, void* d_ws, size_t ws_size,
                              hipStream_t stream) {
    (void)in_sizes; (void)n_in; (void)out_size; (void)ws_size;
    const float* zr   = (const float*)d_in[0];
    const float* zi   = (const float*)d_in[1];
    const int*   prev = (const int*)d_in[2];
    const float* cb   = (const float*)d_in[3];
    const float* adj  = (const float*)d_in[4];
    float* out = (float*)d_out;
    char*  ws  = (char*)d_ws;

    int*      minidx = (int*)(ws + WS_MINIDX);
    unsigned* counts = (unsigned*)(ws + WS_COUNTS);
    double*   loss   = (double*)(ws + WS_LOSS);
    float*    c2     = (float*)(ws + WS_C2);
    uint16_t* cbh    = (uint16_t*)(ws + WS_CBH);
    uint16_t* cbl    = (uint16_t*)(ws + WS_CBL);

    hipMemsetAsync(counts, 0, N_SYMBOLS * sizeof(unsigned), stream);
    hipMemsetAsync(loss, 0, sizeof(double), stream);

    cb_prep<<<N_SYMBOLS, 128, 0, stream>>>(cb, cbh, cbl, c2);
    vq_argmin<<<N_TOKENS / 64, 128, 0, stream>>>(zr, zi, prev, cbh, cbl, c2, adj, minidx);
    vq_output<<<N_TOKENS / 8, 256, 0, stream>>>(zr, zi, cb, minidx, out, loss, counts);
    vq_finalize<<<1, 256, 0, stream>>>(loss, counts, out);
}
